// GRNNTransformSimple_55259049230549
// MI455X (gfx1250) — matrix operand order repaired and checked by reading, not compiled, not run on hardware
//
#include <hip/hip_runtime.h>
#include <hip/hip_bf16.h>

// GRNN tree transform for MI455X (gfx1250), wave32 + WMMA bf16 + async LDS copy.
// Workspace layout (d_ws):
//   [0      .. 32768)        : Wh pre-converted to bf16, transposed+padded [64][200]
//   [32768  .. 268,451,840)  : heap-ordered bf16 embeddings [B=128][16383][64]
//
// children of heap node m are heap nodes 2m+1, 2m+2 -> hL||hR is a contiguous
// 128-element (256 B) run: the "gather" is a straight copy.

#define NN 16383          // nodes per jet
#define HD 64             // hidden
#define KD 192            // 3*H
#define ROWPAD 200        // LDS row stride in ushort (400 B = 25*16 B, conflict-free)
#define WHT_USHORTS (HD * ROWPAD)       // 12800
#define EMB_OFS 16384                   // ushort offset of embAll inside d_ws

typedef __attribute__((ext_vector_type(16))) __bf16 v16bf;
typedef __attribute__((ext_vector_type(8)))  float  v8f;

union FragB {
    uint4 q[2];
    v16bf v;
};

__device__ __forceinline__ unsigned short f2bf(float f) {
    unsigned int u = __builtin_bit_cast(unsigned int, f);
    u += 0x7FFFu + ((u >> 16) & 1u);     // round-to-nearest-even
    return (unsigned short)(u >> 16);
}

// ---------------------------------------------------------------------------
// Prep: Wh [192][64] f32 -> bf16, transposed to [n][k] with ROWPAD padding,
// matching the inner kernel's LDS layout exactly (blocks then flat-copy it).
// ---------------------------------------------------------------------------
__global__ __launch_bounds__(256) void grnn_prep_kernel(
    const float* __restrict__ Wh, unsigned short* __restrict__ WhTg)
{
    const int i = blockIdx.x * 256 + threadIdx.x;    // 0 .. 12287
    if (i < KD * HD) {
        const int k = i >> 6, n = i & 63;
        WhTg[n * ROWPAD + k] = f2bf(Wh[i]);
    }
}

// ---------------------------------------------------------------------------
// Leaf level (i = 13): emb = relu(contents @ Wu + bu), one row per thread.
// ---------------------------------------------------------------------------
__global__ __launch_bounds__(256) void grnn_leaf_kernel(
    const float* __restrict__ contents,
    const float* __restrict__ Wu,
    const float* __restrict__ bu,
    unsigned short* __restrict__ embAll)
{
    __shared__ float sWu[8 * HD];
    __shared__ float sbu[HD];
    const int tid = threadIdx.x;
    for (int i = tid; i < 8 * HD; i += 256) sWu[i] = Wu[i];
    if (tid < HD) sbu[tid] = bu[tid];
    __syncthreads();

    const int row = blockIdx.x * 256 + tid;          // 0 .. 128*8192-1
    const int b = row >> 13;
    const int j = row & 8191;
    const size_t heap = (size_t)b * NN + 8191 + j;

    const float* __restrict__ x = contents + heap * 8;
    float xv[8];
#pragma unroll
    for (int f = 0; f < 8; ++f) xv[f] = x[f];

    unsigned short* __restrict__ dst = embAll + heap * HD;  // 128 B aligned
#pragma unroll
    for (int c0 = 0; c0 < HD; c0 += 8) {
        unsigned short t[8];
#pragma unroll
        for (int c = 0; c < 8; ++c) {
            float acc = sbu[c0 + c];
#pragma unroll
            for (int f = 0; f < 8; ++f) acc += xv[f] * sWu[f * HD + c0 + c];
            acc = acc > 0.f ? acc : 0.f;
            t[c] = f2bf(acc);
        }
        uint4 p;
        p.x = (unsigned)t[0] | ((unsigned)t[1] << 16);
        p.y = (unsigned)t[2] | ((unsigned)t[3] << 16);
        p.z = (unsigned)t[4] | ((unsigned)t[5] << 16);
        p.w = (unsigned)t[6] | ((unsigned)t[7] << 16);
        *(uint4*)(dst + c0) = p;
    }
}

// ---------------------------------------------------------------------------
// Inner levels (i = 12 .. 0):
//   emb_i = relu([hL, hR, relu(x@Wu+bu)] @ Wh + bh)
// Block = 128 threads = 4 waves; each wave processes 8 tiles of 16 rows x 64
// cols; per tile: async global->LDS child staging, VALU u-compute, then
// 24x v_wmma_f32_16x16x32_bf16 (6 k-steps x 4 n-tiles).
// ---------------------------------------------------------------------------
__global__ __launch_bounds__(128) void grnn_inner_kernel(
    const float* __restrict__ contents,
    const float* __restrict__ Wu,
    const float* __restrict__ bu,
    const float* __restrict__ bh,
    const unsigned short* __restrict__ WhTg,   // pre-padded bf16 [64][200]
    unsigned short* __restrict__ embAll,
    int level,
    int numTiles,
    float* __restrict__ rootOut)               // non-null only for level 0
{
    __shared__ unsigned short sWhT[HD][ROWPAD];      // B operand, bf16
    __shared__ unsigned short sH[4][16][ROWPAD];     // per-wave h tile, bf16
    __shared__ float sWu[8 * HD];
    __shared__ float sbu[HD];
    __shared__ float sbh[HD];

    const int tid = threadIdx.x;
    const int n = 1 << level;
    const int off = n - 1;

    // Flat copy of pre-converted, pre-padded WhT (no transpose, no cvt).
    for (int i = tid; i < WHT_USHORTS / 2; i += 128)
        ((unsigned int*)&sWhT[0][0])[i] = ((const unsigned int*)WhTg)[i];
    for (int i = tid; i < 8 * HD; i += 128) sWu[i] = Wu[i];
    if (tid < HD) { sbu[tid] = bu[tid]; sbh[tid] = bh[tid]; }
    __syncthreads();

    const int wave = tid >> 5;
    const int lane = tid & 31;
    const int l15 = lane & 15;
    const int lhalf = lane >> 4;

    for (int tt = 0; tt < 8; ++tt) {
        const int tile = blockIdx.x * 32 + tt * 4 + wave;   // numTiles % 4 == 0
        if (tile >= numTiles) break;                         // wave-uniform
        const int rowBase = tile * 16;

        // --- async stage children: h[:,0:128] = emb[2m+1] || emb[2m+2] ---
#pragma unroll 4
        for (int rr = 0; rr < 16; ++rr) {
            const int row = rowBase + rr;
            const int b = row >> level;
            const int j = row & (n - 1);
            const size_t childHeap = (size_t)b * NN + 2 * (size_t)(off + j) + 1;
            const unsigned short* __restrict__ src =
                embAll + childHeap * HD + lane * 4;          // 8 B per lane
            const unsigned long long gaddr = (unsigned long long)(uintptr_t)src;
            const unsigned int ldsOff =
                (unsigned int)(uintptr_t)&sH[wave][rr][lane * 4];
            asm volatile("global_load_async_to_lds_b64 %0, %1, off"
                         :: "v"(ldsOff), "v"(gaddr) : "memory");
            __builtin_prefetch(contents + ((size_t)b * NN + off + j) * 8, 0, 0);
        }

        // --- u = relu(x@Wu+bu) into h[:,128:192] (overlaps async copy) ---
#pragma unroll 4
        for (int rr = 0; rr < 16; ++rr) {
            const int row = rowBase + rr;
            const int b = row >> level;
            const int j = row & (n - 1);
            const float* __restrict__ x =
                contents + ((size_t)b * NN + off + j) * 8;
            const float4 x0 = *(const float4*)x;
            const float4 x1 = *(const float4*)(x + 4);
            const float xv[8] = {x0.x, x0.y, x0.z, x0.w, x1.x, x1.y, x1.z, x1.w};
            const int c0 = lane * 2;
            float a0 = sbu[c0], a1 = sbu[c0 + 1];
#pragma unroll
            for (int f = 0; f < 8; ++f) {
                a0 += xv[f] * sWu[f * HD + c0];
                a1 += xv[f] * sWu[f * HD + c0 + 1];
            }
            a0 = a0 > 0.f ? a0 : 0.f;
            a1 = a1 > 0.f ? a1 : 0.f;
            const unsigned int packed =
                (unsigned)f2bf(a0) | ((unsigned)f2bf(a1) << 16);
            *(unsigned int*)&sH[wave][rr][128 + c0] = packed;
        }

        // Children must be resident in LDS before A-fragment loads.
        asm volatile("s_wait_asynccnt 0" ::: "memory");

        // --- WMMA: D = h[16x192] @ Wh[192x64], f32 accum seeded with bh ---
        v8f acc[4];
#pragma unroll
        for (int nt = 0; nt < 4; ++nt) {
            const float bv = sbh[nt * 16 + l15];
#pragma unroll
            for (int r = 0; r < 8; ++r) acc[nt][r] = bv;
        }

#pragma unroll
        for (int kt = 0; kt < 6; ++kt) {
            const int kb = kt * 32;
            FragB A;   // lanes 0-15: K kb+0..7 / kb+16..23; lanes 16-31: +8
            A.q[0] = *(const uint4*)&sH[wave][l15][kb + lhalf * 8];
            A.q[1] = *(const uint4*)&sH[wave][l15][kb + 16 + lhalf * 8];
#pragma unroll
            for (int nt = 0; nt < 4; ++nt) {
                FragB Bf;  // lane holds column N = nt*16+l15, K kb+0..7 / kb+16..23; lanes 16-31: +8
                const unsigned short* bp = &sWhT[nt * 16 + l15][kb + lhalf * 8];
                Bf.q[0] = *(const uint4*)bp;
                Bf.q[1] = *(const uint4*)(bp + 16);
                acc[nt] = __builtin_amdgcn_wmma_f32_16x16x32_bf16(
                    false, A.v, false, Bf.v, (short)0, acc[nt], false, false);
            }
        }

        // --- epilogue: relu + store ---
#pragma unroll
        for (int r = 0; r < 8; ++r) {
            const int m = rowBase + r + lhalf * 8;
            const int b = m >> level;
            const int j = m & (n - 1);
            if (rootOut) {
#pragma unroll
                for (int nt = 0; nt < 4; ++nt) {
                    float v = acc[nt][r];
                    rootOut[(size_t)m * HD + nt * 16 + l15] = v > 0.f ? v : 0.f;
                }
            } else {
                unsigned short* __restrict__ dst =
                    embAll + ((size_t)b * NN + off + j) * HD + l15;
#pragma unroll
                for (int nt = 0; nt < 4; ++nt) {
                    float v = acc[nt][r];
                    dst[nt * 16] = f2bf(v > 0.f ? v : 0.f);
                }
            }
        }
    }
}

// ---------------------------------------------------------------------------
extern "C" void kernel_launch(void* const* d_in, const int* in_sizes, int n_in,
                              void* d_out, int out_size, void* d_ws, size_t ws_size,
                              hipStream_t stream) {
    const float* contents = (const float*)d_in[0];
    // d_in[1] = children (int) — unused: heap children are implicit (2m+1, 2m+2)
    const float* Wu = (const float*)d_in[2];
    const float* bu = (const float*)d_in[3];
    const float* Wh = (const float*)d_in[4];
    const float* bh = (const float*)d_in[5];
    float* out = (float*)d_out;

    unsigned short* WhTg   = (unsigned short*)d_ws;           // 12800 ushorts
    unsigned short* embAll = (unsigned short*)d_ws + EMB_OFS; // [128][16383][64]

    // Pre-convert Wh to bf16 in the exact (transposed, padded) LDS layout.
    grnn_prep_kernel<<<48, 256, 0, stream>>>(Wh, WhTg);

    // Leaf level 13: 128*8192 rows, one per thread.
    grnn_leaf_kernel<<<4096, 256, 0, stream>>>(contents, Wu, bu, embAll);

    // Inner levels 12 .. 0. numTiles = 128*2^level / 16 = 8 << level.
    for (int level = 12; level >= 0; --level) {
        const int numTiles = 8 << level;
        const int blocks = (numTiles + 31) / 32;
        grnn_inner_kernel<<<blocks, 128, 0, stream>>>(
            contents, Wu, bu, bh, WhTg, embAll, level, numTiles,
            (level == 0) ? out : nullptr);
    }
}
